// SingleHeadAttention_FA_1786706395422
// MI455X (gfx1250) — compile-verified, matrix operand order checked by reading, not run on hardware
//
#include <hip/hip_runtime.h>
#include <hip/hip_bf16.h>

#define B_ 8
#define L_ 1024
#define D_ 256
#define F_ 11
#define LT_ (L_/16)
#define DT_ (D_/16)
#define INV_TEMP 0.0625f   // 1/sqrt(256)
#define LN_EPS 1e-6f

typedef __attribute__((ext_vector_type(16))) _Float16 v16h;
typedef __attribute__((ext_vector_type(8)))  _Float16 v8h;
typedef __attribute__((ext_vector_type(8)))  float    v8f;

// ---------------------------------------------------------------- WMMA core
__device__ __forceinline__ v8f wmma16(v16h a, v16h b, v8f c) {
  // D = A(16x32 f16) x B(32x16 f16) + C(16x16 f32)
  return __builtin_amdgcn_wmma_f32_16x16x32_f16(
      /*neg_a=*/false, a, /*neg_b=*/false, b,
      /*c_mod=*/(short)0, c, /*reuse_a=*/false, /*reuse_b=*/false);
}

// 16x32 fragment (ISA 16-bit A layout) from row-major f32, converted to f16.
// lanes 0-15: row=lane, K in {0..7}U{16..23}; lanes 16-31: row=lane-16,
// K in {8..15}U{24..31}.
__device__ __forceinline__ v16h ld_frag_f32(const float* __restrict__ base, int ld) {
  const int lane = threadIdx.x & 31;
  const int row  = lane & 15;
  const int kb   = (lane & 16) ? 8 : 0;
  const float* p = base + (size_t)row * ld + kb;
  float4 x0 = *(const float4*)(p + 0);
  float4 x1 = *(const float4*)(p + 4);
  float4 x2 = *(const float4*)(p + 16);
  float4 x3 = *(const float4*)(p + 20);
  v16h f;
  f[0]=(_Float16)x0.x;  f[1]=(_Float16)x0.y;  f[2]=(_Float16)x0.z;  f[3]=(_Float16)x0.w;
  f[4]=(_Float16)x1.x;  f[5]=(_Float16)x1.y;  f[6]=(_Float16)x1.z;  f[7]=(_Float16)x1.w;
  f[8]=(_Float16)x2.x;  f[9]=(_Float16)x2.y;  f[10]=(_Float16)x2.z; f[11]=(_Float16)x2.w;
  f[12]=(_Float16)x3.x; f[13]=(_Float16)x3.y; f[14]=(_Float16)x3.z; f[15]=(_Float16)x3.w;
  return f;
}

// Same fragment from row-major f16 (two b128 loads per lane).
__device__ __forceinline__ v16h ld_frag_f16(const _Float16* __restrict__ base, int ld) {
  const int lane = threadIdx.x & 31;
  const int row  = lane & 15;
  const int kb   = (lane & 16) ? 8 : 0;
  const _Float16* p = base + (size_t)row * ld + kb;
  v8h a = *(const v8h*)(p);
  v8h b = *(const v8h*)(p + 16);
  v16h f;
#pragma unroll
  for (int e = 0; e < 8; ++e) { f[e] = a[e]; f[8 + e] = b[e]; }
  return f;
}

// Store a 16x16 f32 accumulator (ISA C/D layout) as f16 row-major.
__device__ __forceinline__ void st_tile_f16(v8f acc, _Float16* __restrict__ dst, int ld) {
  const int lane = threadIdx.x & 31;
  const int n  = lane & 15;
  const int m0 = (lane & 16) ? 8 : 0;
#pragma unroll
  for (int r = 0; r < 8; ++r) dst[(size_t)(m0 + r) * ld + n] = (_Float16)acc[r];
}

// ---------------------------------------------------------------- kernel 1:
// 16x64 tile per wave: qp,kp (row-major f16); vp,q2,k2 (transposed f16 = NxK)
__global__ void proj_kernel(const float* __restrict__ q, const float* __restrict__ k,
                            const float* __restrict__ v,
                            const float* __restrict__ w_qs, const float* __restrict__ w_ks,
                            const float* __restrict__ w_vs, const float* __restrict__ w_qs2,
                            const float* __restrict__ w_ks2,
                            _Float16* __restrict__ qph, _Float16* __restrict__ kph,
                            _Float16* __restrict__ vpT, _Float16* __restrict__ q2T,
                            _Float16* __restrict__ k2T) {
  const int wave = (int)((blockIdx.x * blockDim.x + threadIdx.x) >> 5);
  const int ot4 = wave & 3;                    // 64-wide N block
  int r = wave >> 2;
  const int lt = r & (LT_ - 1); r >>= 6;
  const int b  = r & (B_ - 1);
  const int which = r >> 3;                    // 0..4, wave-uniform
  const float* A; const float* W;
  if      (which == 0) { A = q; W = w_qs;  }
  else if (which == 1) { A = k; W = w_ks;  }
  else if (which == 2) { A = v; W = w_vs;  }
  else if (which == 3) { A = q; W = w_qs2; }
  else                 { A = k; W = w_ks2; }
  const float* Ab = A + ((size_t)b * L_ + lt * 16) * D_;
  const float* Wb = W + (size_t)ot4 * 64 * D_;           // W is (o,i) = N x K
  v8f acc[4] = {{}, {}, {}, {}};
#pragma unroll
  for (int k0 = 0; k0 < D_; k0 += 32) {
    v16h af = ld_frag_f32(Ab + k0, D_);
#pragma unroll
    for (int s = 0; s < 4; ++s)
      acc[s] = wmma16(af, ld_frag_f32(Wb + (size_t)s * 16 * D_ + k0, D_), acc[s]);
  }
  const int lane = threadIdx.x & 31;
  const int n  = lane & 15;
  const int m0 = (lane & 16) ? 8 : 0;
  if (which < 2) {
    _Float16* dst = (which == 0 ? qph : kph) + ((size_t)b * L_ + lt * 16) * D_ + ot4 * 64;
#pragma unroll
    for (int s = 0; s < 4; ++s)
#pragma unroll
      for (int rr = 0; rr < 8; ++rr)
        dst[(size_t)(m0 + rr) * D_ + s * 16 + n] = (_Float16)acc[s][rr];
  } else {
    _Float16* dst = (which == 2 ? vpT : (which == 3 ? q2T : k2T))
                    + ((size_t)b * D_ + ot4 * 64) * L_ + lt * 16;
#pragma unroll
    for (int s = 0; s < 4; ++s)
#pragma unroll
      for (int rr = 0; rr < 8; ++rr)
        dst[(size_t)(s * 16 + n) * L_ + (m0 + rr)] = (_Float16)acc[s][rr];
  }
}

// ---------------------------------------------------------------- kernel 2:
// fa[b,i,j] = softmax_j( mask_j( sum_f imp[f]*|x[i,f]-x[j,f]| ) )   (f16 out)
__global__ void fa_kernel(const float* __restrict__ x, const float* __restrict__ imp,
                          const int* __restrict__ lens, _Float16* __restrict__ fah) {
  __shared__ float s[L_];
  __shared__ float red[256];
  __shared__ float xi[16];
  __shared__ float impS[16];
  const int b = (int)(blockIdx.x >> 10);
  const int i = (int)(blockIdx.x & (L_ - 1));
  const int t = threadIdx.x;
  if (t < F_) { xi[t] = x[((size_t)b * L_ + i) * F_ + t]; impS[t] = imp[t]; }
  __syncthreads();
  const int len = lens[b];
  float mx = -3.4e38f;
  for (int j = t; j < L_; j += 256) {
    const float* xj = x + ((size_t)b * L_ + j) * F_;
    float acc = 0.f;
#pragma unroll
    for (int f = 0; f < F_; ++f) acc += impS[f] * fabsf(xi[f] - xj[f]);
    if (j >= len) acc = -__builtin_inff();
    s[j] = acc;
    mx = fmaxf(mx, acc);
  }
  red[t] = mx; __syncthreads();
  for (int st = 128; st > 0; st >>= 1) { if (t < st) red[t] = fmaxf(red[t], red[t + st]); __syncthreads(); }
  const float rowmax = red[0];
  __syncthreads();
  float sum = 0.f;
  for (int j = t; j < L_; j += 256) { float e = expf(s[j] - rowmax); s[j] = e; sum += e; }
  red[t] = sum; __syncthreads();
  for (int st = 128; st > 0; st >>= 1) { if (t < st) red[t] += red[t + st]; __syncthreads(); }
  const float inv = 1.f / red[0];
  _Float16* dst = fah + ((size_t)b * L_ + i) * L_;
  for (int j = t; j < L_; j += 256) dst[j] = (_Float16)(s[j] * inv);
}

// ---------------------------------------------------------------- kernel 3:
// q2f = fa @ q2 ; k2f = fa @ k2   (K = 1024; 16x64 tile per wave)
__global__ void favg_kernel(const _Float16* __restrict__ fah, const _Float16* __restrict__ q2T,
                            const _Float16* __restrict__ k2T,
                            _Float16* __restrict__ q2fh, _Float16* __restrict__ k2fh) {
  const int wave = (int)((blockIdx.x * blockDim.x + threadIdx.x) >> 5);
  const int dt4 = wave & 3;
  int r = wave >> 2;
  const int it = r & (LT_ - 1); r >>= 6;
  const int b  = r & (B_ - 1);
  const int which = r >> 3;                 // 0 -> q2f, 1 -> k2f
  const _Float16* Ab = fah + ((size_t)b * L_ + it * 16) * L_;
  const _Float16* Wb = (which ? k2T : q2T) + ((size_t)b * D_ + dt4 * 64) * L_;
  v8f acc[4] = {{}, {}, {}, {}};
  for (int k0 = 0; k0 < L_; k0 += 32) {
    __builtin_prefetch(Ab + k0 + 128, 0, 1);
    __builtin_prefetch(Wb + k0 + 128, 0, 1);
    v16h af = ld_frag_f16(Ab + k0, L_);
#pragma unroll
    for (int s = 0; s < 4; ++s)
      acc[s] = wmma16(af, ld_frag_f16(Wb + (size_t)s * 16 * L_ + k0, L_), acc[s]);
  }
  _Float16* dst = (which ? k2fh : q2fh) + ((size_t)b * L_ + it * 16) * D_ + dt4 * 64;
#pragma unroll
  for (int s = 0; s < 4; ++s) st_tile_f16(acc[s], dst + s * 16, D_);
}

// ---------------------------------------------------------------- kernel 4:
// attn = tanh( mask_j * (qp.kp^T + q2f.k2f^T) / sqrt(D) )
// writes f32 output + f16 copy for the attn@vp GEMM; 16x64 tile per wave
__global__ void attn_kernel(const _Float16* __restrict__ qph, const _Float16* __restrict__ kph,
                            const _Float16* __restrict__ q2fh, const _Float16* __restrict__ k2fh,
                            const int* __restrict__ lens, float* __restrict__ attn_out,
                            _Float16* __restrict__ attnh) {
  const int wave = (int)((blockIdx.x * blockDim.x + threadIdx.x) >> 5);
  const int jt4 = wave & 15;                 // 64-wide column block
  const int it  = (wave >> 4) & (LT_ - 1);
  const int b   = wave >> 10;
  const _Float16* A1 = qph  + ((size_t)b * L_ + it * 16) * D_;
  const _Float16* A2 = q2fh + ((size_t)b * L_ + it * 16) * D_;
  const _Float16* B1 = kph  + ((size_t)b * L_ + jt4 * 64) * D_;
  const _Float16* B2 = k2fh + ((size_t)b * L_ + jt4 * 64) * D_;
  v8f acc[4] = {{}, {}, {}, {}};
#pragma unroll
  for (int k0 = 0; k0 < D_; k0 += 32) {
    v16h a1 = ld_frag_f16(A1 + k0, D_);
    v16h a2 = ld_frag_f16(A2 + k0, D_);
#pragma unroll
    for (int s = 0; s < 4; ++s) {
      acc[s] = wmma16(a1, ld_frag_f16(B1 + (size_t)s * 16 * D_ + k0, D_), acc[s]);
      acc[s] = wmma16(a2, ld_frag_f16(B2 + (size_t)s * 16 * D_ + k0, D_), acc[s]);
    }
  }
  const int len  = lens[b];
  const int lane = threadIdx.x & 31;
  const int n  = lane & 15;
  const int m0 = (lane & 16) ? 8 : 0;
  float*    dst  = attn_out + ((size_t)b * L_ + it * 16) * L_ + jt4 * 64;
  _Float16* dsth = attnh    + ((size_t)b * L_ + it * 16) * L_ + jt4 * 64;
#pragma unroll
  for (int s = 0; s < 4; ++s) {
    const int j = jt4 * 64 + s * 16 + n;
#pragma unroll
    for (int rr = 0; rr < 8; ++rr) {
      float val = (j < len) ? tanhf(acc[s][rr] * INV_TEMP) : 0.f;
      dst [(size_t)(m0 + rr) * L_ + s * 16 + n] = val;
      dsth[(size_t)(m0 + rr) * L_ + s * 16 + n] = (_Float16)val;
    }
  }
}

// ---------------------------------------------------------------- kernel 5:
// o = attn @ vp  (K = 1024; f16 attn copy, 16x64 tile per wave)
__global__ void ov_kernel(const _Float16* __restrict__ attnh, const _Float16* __restrict__ vpT,
                          _Float16* __restrict__ oh) {
  const int wave = (int)((blockIdx.x * blockDim.x + threadIdx.x) >> 5);
  const int dt4 = wave & 3;
  const int it  = (wave >> 2) & (LT_ - 1);
  const int b   = wave >> 8;
  const _Float16* Ab = attnh + ((size_t)b * L_ + it * 16) * L_;
  const _Float16* Wb = vpT   + ((size_t)b * D_ + dt4 * 64) * L_;
  v8f acc[4] = {{}, {}, {}, {}};
  for (int k0 = 0; k0 < L_; k0 += 32) {
    __builtin_prefetch(Ab + k0 + 128, 0, 1);
    __builtin_prefetch(Wb + k0 + 128, 0, 1);
    v16h af = ld_frag_f16(Ab + k0, L_);
#pragma unroll
    for (int s = 0; s < 4; ++s)
      acc[s] = wmma16(af, ld_frag_f16(Wb + (size_t)s * 16 * L_ + k0, L_), acc[s]);
  }
  _Float16* dst = oh + ((size_t)b * L_ + it * 16) * D_ + dt4 * 64;
#pragma unroll
  for (int s = 0; s < 4; ++s) st_tile_f16(acc[s], dst + s * 16, D_);
}

// ---------------------------------------------------------------- kernel 6:
// t = o @ w_fc^T + q   (f32, pre-LayerNorm; 16x64 tile per wave)
__global__ void fc_kernel(const _Float16* __restrict__ oh, const float* __restrict__ w_fc,
                          const float* __restrict__ q, float* __restrict__ tbuf) {
  const int wave = (int)((blockIdx.x * blockDim.x + threadIdx.x) >> 5);
  const int ot4 = wave & 3;
  const int it  = (wave >> 2) & (LT_ - 1);
  const int b   = wave >> 8;
  const _Float16* Ab = oh   + ((size_t)b * L_ + it * 16) * D_;
  const float*    Wb = w_fc + (size_t)ot4 * 64 * D_;
  v8f acc[4] = {{}, {}, {}, {}};
#pragma unroll
  for (int k0 = 0; k0 < D_; k0 += 32) {
    v16h af = ld_frag_f16(Ab + k0, D_);
#pragma unroll
    for (int s = 0; s < 4; ++s)
      acc[s] = wmma16(af, ld_frag_f32(Wb + (size_t)s * 16 * D_ + k0, D_), acc[s]);
  }
  const int lane = threadIdx.x & 31;
  const int n  = lane & 15;
  const int m0 = (lane & 16) ? 8 : 0;
  const float* qb = q    + ((size_t)b * L_ + it * 16) * D_ + ot4 * 64;
  float*       db = tbuf + ((size_t)b * L_ + it * 16) * D_ + ot4 * 64;
#pragma unroll
  for (int s = 0; s < 4; ++s)
#pragma unroll
    for (int rr = 0; rr < 8; ++rr)
      db[(size_t)(m0 + rr) * D_ + s * 16 + n] =
          acc[s][rr] + qb[(size_t)(m0 + rr) * D_ + s * 16 + n];
}

// ---------------------------------------------------------------- kernel 7:
// LayerNorm over D per row
__global__ void ln_kernel(const float* __restrict__ tbuf, const float* __restrict__ gamma,
                          const float* __restrict__ beta, float* __restrict__ out) {
  __shared__ float rs[256];
  __shared__ float rq[256];
  const size_t row = blockIdx.x;
  const int t = threadIdx.x;
  const float vv = tbuf[row * D_ + t];
  rs[t] = vv; rq[t] = vv * vv;
  __syncthreads();
  for (int st = 128; st > 0; st >>= 1) {
    if (t < st) { rs[t] += rs[t + st]; rq[t] += rq[t + st]; }
    __syncthreads();
  }
  const float mu  = rs[0] * (1.f / D_);
  const float var = rq[0] * (1.f / D_) - mu * mu;
  out[row * D_ + t] = (vv - mu) * rsqrtf(var + LN_EPS) * gamma[t] + beta[t];
}

// ---------------------------------------------------------------- launch
extern "C" void kernel_launch(void* const* d_in, const int* in_sizes, int n_in,
                              void* d_out, int out_size, void* d_ws, size_t ws_size,
                              hipStream_t stream) {
  const float* q     = (const float*)d_in[0];
  const float* k     = (const float*)d_in[1];
  const float* v     = (const float*)d_in[2];
  const float* x     = (const float*)d_in[3];
  const int*   lens  = (const int*)d_in[4];
  const float* w_qs  = (const float*)d_in[5];
  const float* w_ks  = (const float*)d_in[6];
  const float* w_vs  = (const float*)d_in[7];
  const float* w_qs2 = (const float*)d_in[8];
  const float* w_ks2 = (const float*)d_in[9];
  const float* w_fc  = (const float*)d_in[10];
  const float* fimp  = (const float*)d_in[11];
  const float* gamma = (const float*)d_in[12];
  const float* beta  = (const float*)d_in[13];

  float* out_f  = (float*)d_out;                          // (B,L,D)
  float* attn_f = out_f + (size_t)B_ * L_ * D_;           // (B,L,L)

  char* ws = (char*)d_ws;
  const size_t SZH  = (size_t)B_ * L_ * D_ * sizeof(_Float16);   // 4 MiB each
  const size_t SZLL = (size_t)B_ * L_ * L_ * sizeof(_Float16);   // 16 MiB
  _Float16* qph  = (_Float16*)(ws + 0 * SZH);
  _Float16* kph  = (_Float16*)(ws + 1 * SZH);
  _Float16* vpT  = (_Float16*)(ws + 2 * SZH);
  _Float16* q2T  = (_Float16*)(ws + 3 * SZH);
  _Float16* k2T  = (_Float16*)(ws + 4 * SZH);
  _Float16* q2fh = (_Float16*)(ws + 5 * SZH);
  _Float16* k2fh = (_Float16*)(ws + 6 * SZH);
  _Float16* oh   = (_Float16*)(ws + 7 * SZH);
  _Float16* fah  = (_Float16*)(ws + 8 * SZH);                    // 16 MiB
  _Float16* attnh= (_Float16*)(ws + 8 * SZH + SZLL);             // 16 MiB
  float*    tbuf = (float*)(ws + 8 * SZH + 2 * SZLL);            // 8 MiB

  proj_kernel<<<1280, 256, 0, stream>>>(q, k, v, w_qs, w_ks, w_vs, w_qs2, w_ks2,
                                        qph, kph, vpT, q2T, k2T);
  fa_kernel<<<B_ * L_, 256, 0, stream>>>(x, fimp, lens, fah);
  favg_kernel<<<512, 256, 0, stream>>>(fah, q2T, k2T, q2fh, k2fh);
  attn_kernel<<<1024, 256, 0, stream>>>(qph, kph, q2fh, k2fh, lens, attn_f, attnh);
  ov_kernel<<<256, 256, 0, stream>>>(attnh, vpT, oh);
  fc_kernel<<<256, 256, 0, stream>>>(oh, w_fc, q, tbuf);
  ln_kernel<<<B_ * L_, 256, 0, stream>>>(tbuf, gamma, beta, out_f);
}
